// SelfAttention_74156905333034
// MI455X (gfx1250) — matrix operand order repaired and checked by reading, not compiled, not run on hardware
//
#include <hip/hip_runtime.h>

typedef __attribute__((ext_vector_type(16))) __bf16 v16bf;
typedef __attribute__((ext_vector_type(8)))  float  v8f;
typedef __attribute__((ext_vector_type(4)))  int    v4i;

#define BK   32
#define TSTR 40   // LDS row stride in bf16 elements: 80B rows, 16B-aligned

#if __has_builtin(__builtin_amdgcn_global_load_async_to_lds_b128) && \
    __has_builtin(__builtin_amdgcn_s_wait_asynccnt)
#define HAVE_ASYNC 1
#else
#define HAVE_ASYNC 0
#endif

#if HAVE_ASYNC
__device__ __forceinline__ void async_cp16(void* lds, const void* g) {
    __builtin_amdgcn_global_load_async_to_lds_b128(
        (__attribute__((address_space(1))) v4i*)(void*)g,
        (__attribute__((address_space(3))) v4i*)lds,
        /*offset=*/0, /*cpol=*/0);
}
#endif

__device__ __forceinline__ v16bf load_frag16(const __bf16* p0, const __bf16* p1) {
    union { uint4 u[2]; v16bf v; } U;
    U.u[0] = *(const uint4*)p0;
    U.u[1] = *(const uint4*)p1;
    return U.v;
}

// C[m,n] = sum_k A[m,k] * B(k,n)
//   A row-major [M,K] (f32 or bf16)
//   B: NT -> B(k,n) = Brow[n*ldb + k] (row-major [N,K]); NN -> B(k,n) = B[k*ldb + n]
// OUT_MODE: 0 = bf16*scale   1 = f32 + causal mask (n>m -> -3e38)   2 = f32
template<bool A_F32, bool B_F32, bool B_NN, int OUT_MODE,
         int BM, int BN, int WM, int WN>
__global__ __launch_bounds__(256)
void gemm_wmma(const void* __restrict__ Ain, const void* __restrict__ Bin,
               void* __restrict__ Cout,
               int K, int lda, int ldb, int ldc,
               size_t aBatch, size_t bBatch, size_t cBatch, float scale)
{
    constexpr int  MI    = WM / 16;
    constexpr int  NI    = WN / 16;
    constexpr int  WGN   = BN / WN;              // waves along N (BM/WM * BN/WN == 8)
    constexpr bool A_ASY = (HAVE_ASYNC != 0) && !A_F32;
    constexpr bool B_ASY = (HAVE_ASYNC != 0) && !B_F32 && !B_NN;
    constexpr bool ANY_ASY = A_ASY || B_ASY;

    __shared__ __align__(16) __bf16 aT[2][BM * TSTR];
    __shared__ __align__(16) __bf16 bT[2][BN * TSTR];

    const int tid  = threadIdx.x;
    const int lane = tid & 31;
    const int wave = tid >> 5;
    const int wM   = wave / WGN;
    const int wN   = wave % WGN;
    const int l16  = lane & 15;
    const int lh   = lane >> 4;

    const int nBase = blockIdx.x * BN;
    const int mBase = blockIdx.y * BM;
    const int bz    = blockIdx.z;

    if (OUT_MODE == 1 && nBase > mBase + (BM - 1)) return;  // fully above diagonal

    const float*  Af = (const float*) Ain + aBatch * bz;
    const __bf16* Ab = (const __bf16*)Ain + aBatch * bz;
    const float*  Bf = (const float*) Bin + bBatch * bz;
    const __bf16* Bb = (const __bf16*)Bin + bBatch * bz;

    // register prefetch buffers (unused paths are eliminated)
    float4   aRf[BM / 32];
    uint4    aRu[BM / 64];
    float4   bRf[BN / 32];
    uint4    bRu[BN / 64];
    unsigned bRn[BN / 16];

    auto loadRegs = [&](int kb) {
        if (A_F32) {
#pragma unroll
            for (int p = 0; p < BM / 32; ++p) {
                int r = (tid >> 3) + p * 32, c = (tid & 7) * 4;
                aRf[p] = *(const float4*)(Af + (size_t)(mBase + r) * lda + kb + c);
            }
        } else if (!A_ASY) {
#pragma unroll
            for (int p = 0; p < BM / 64; ++p) {
                int r = (tid >> 2) + p * 64, c = (tid & 3) * 8;
                aRu[p] = *(const uint4*)(Ab + (size_t)(mBase + r) * lda + kb + c);
            }
        }
        if (B_NN) {
#pragma unroll
            for (int p = 0; p < BN / 16; ++p) {
                int idx = p * 256 + tid;
                int k = idx / (BN / 2), n2 = (idx % (BN / 2)) * 2;
                bRn[p] = *(const unsigned*)(Bb + (size_t)(kb + k) * ldb + nBase + n2);
            }
        } else if (B_F32) {
#pragma unroll
            for (int p = 0; p < BN / 32; ++p) {
                int r = (tid >> 3) + p * 32, c = (tid & 7) * 4;
                bRf[p] = *(const float4*)(Bf + (size_t)(nBase + r) * ldb + kb + c);
            }
        } else if (!B_ASY) {
#pragma unroll
            for (int p = 0; p < BN / 64; ++p) {
                int r = (tid >> 2) + p * 64, c = (tid & 3) * 8;
                bRu[p] = *(const uint4*)(Bb + (size_t)(nBase + r) * ldb + kb + c);
            }
        }
    };

    auto storeLds = [&](int buf) {
        if (A_F32) {
#pragma unroll
            for (int p = 0; p < BM / 32; ++p) {
                int r = (tid >> 3) + p * 32, c = (tid & 7) * 4;
                __bf16* d = &aT[buf][r * TSTR + c];
                d[0] = (__bf16)aRf[p].x; d[1] = (__bf16)aRf[p].y;
                d[2] = (__bf16)aRf[p].z; d[3] = (__bf16)aRf[p].w;
            }
        } else if (!A_ASY) {
#pragma unroll
            for (int p = 0; p < BM / 64; ++p) {
                int r = (tid >> 2) + p * 64, c = (tid & 3) * 8;
                *(uint4*)&aT[buf][r * TSTR + c] = aRu[p];
            }
        }
        if (B_NN) {
#pragma unroll
            for (int p = 0; p < BN / 16; ++p) {
                int idx = p * 256 + tid;
                int k = idx / (BN / 2), n2 = (idx % (BN / 2)) * 2;
                unsigned u = bRn[p];
                bT[buf][(n2 + 0) * TSTR + k] = ((const __bf16*)&u)[0];
                bT[buf][(n2 + 1) * TSTR + k] = ((const __bf16*)&u)[1];
            }
        } else if (B_F32) {
#pragma unroll
            for (int p = 0; p < BN / 32; ++p) {
                int r = (tid >> 3) + p * 32, c = (tid & 7) * 4;
                __bf16* d = &bT[buf][r * TSTR + c];
                d[0] = (__bf16)bRf[p].x; d[1] = (__bf16)bRf[p].y;
                d[2] = (__bf16)bRf[p].z; d[3] = (__bf16)bRf[p].w;
            }
        } else if (!B_ASY) {
#pragma unroll
            for (int p = 0; p < BN / 64; ++p) {
                int r = (tid >> 2) + p * 64, c = (tid & 3) * 8;
                *(uint4*)&bT[buf][r * TSTR + c] = bRu[p];
            }
        }
    };

#if HAVE_ASYNC
    auto issueAsync = [&](int buf, int kb) {
        if (A_ASY) {
#pragma unroll
            for (int p = 0; p < BM / 64; ++p) {
                int r = (tid >> 2) + p * 64, c = (tid & 3) * 8;
                async_cp16(&aT[buf][r * TSTR + c],
                           Ab + (size_t)(mBase + r) * lda + kb + c);
            }
        }
        if (B_ASY) {
#pragma unroll
            for (int p = 0; p < BN / 64; ++p) {
                int r = (tid >> 2) + p * 64, c = (tid & 3) * 8;
                async_cp16(&bT[buf][r * TSTR + c],
                           Bb + (size_t)(nBase + r) * ldb + kb + c);
            }
        }
    };
#endif

    v8f acc[MI][NI];
    v8f z = {0.f,0.f,0.f,0.f,0.f,0.f,0.f,0.f};
#pragma unroll
    for (int mi = 0; mi < MI; ++mi)
#pragma unroll
        for (int ni = 0; ni < NI; ++ni) acc[mi][ni] = z;

    const int nIter = K / BK;

    // prologue: stage k-tile 0 into buffer 0
    loadRegs(0);
    storeLds(0);
#if HAVE_ASYNC
    if (ANY_ASY) issueAsync(0, 0);
#endif

    for (int i = 0; i < nIter; ++i) {
        const int  buf     = i & 1;
        const bool hasNext = (i + 1 < nIter);

        if (hasNext) loadRegs((i + 1) * BK);      // global->regs, overlaps compute
#if HAVE_ASYNC
        if (ANY_ASY) __builtin_amdgcn_s_wait_asynccnt(0);  // buf's async writes done
#endif
        __syncthreads();                           // buf visible; old readers retired
#if HAVE_ASYNC
        if (ANY_ASY && hasNext) issueAsync(buf ^ 1, (i + 1) * BK);
#endif

        v16bf af[MI], bfv[NI];
#pragma unroll
        for (int mi = 0; mi < MI; ++mi) {
            const __bf16* p = &aT[buf][(wM * WM + mi * 16 + l16) * TSTR + lh * 8];
            af[mi] = load_frag16(p, p + 16);       // K 0..15 / 16..31 halves
        }
#pragma unroll
        for (int ni = 0; ni < NI; ++ni) {
            const __bf16* p = &bT[buf][(wN * WN + ni * 16 + l16) * TSTR + lh * 8];
            bfv[ni] = load_frag16(p, p + 16);
        }
#pragma unroll
        for (int mi = 0; mi < MI; ++mi)
#pragma unroll
            for (int ni = 0; ni < NI; ++ni)
                acc[mi][ni] = __builtin_amdgcn_wmma_f32_16x16x32_bf16(
                    false, af[mi], false, bfv[ni], (short)0, acc[mi][ni], false, false);

        if (hasNext) {
            __syncthreads();
            storeLds(buf ^ 1);                     // reg-path tiles into next buffer
        }
    }

#pragma unroll
    for (int mi = 0; mi < MI; ++mi)
#pragma unroll
        for (int ni = 0; ni < NI; ++ni)
#pragma unroll
            for (int v = 0; v < 8; ++v) {
                int m = mBase + wM * WM + mi * 16 + lh * 8 + v;
                int n = nBase + wN * WN + ni * 16 + l16;
                float val = acc[mi][ni][v];
                if (OUT_MODE == 0) {
                    __bf16* C = (__bf16*)Cout + cBatch * bz;
                    C[(size_t)m * ldc + n] = (__bf16)(val * scale);
                } else {
                    float* C = (float*)Cout + cBatch * bz;
                    if (OUT_MODE == 1 && n > m) val = -3.0e38f;
                    C[(size_t)m * ldc + n] = val;
                }
            }
}

// One block (256 threads) per query row: causal softmax, f32 logits -> bf16 probs
__global__ __launch_bounds__(256)
void softmax_causal(const float* __restrict__ S, __bf16* __restrict__ P, int T)
{
    const int row = blockIdx.x;           // b*T + t
    const int t   = row % T;
    const int tid = threadIdx.x;
    const float* Srow = S + (size_t)row * T;
    __bf16*      Prow = P + (size_t)row * T;
    __shared__ float red[256];

    float mx = -3.0e38f;
    for (int s = tid; s <= t; s += 256) mx = fmaxf(mx, Srow[s]);
    red[tid] = mx; __syncthreads();
    for (int off = 128; off > 0; off >>= 1) {
        if (tid < off) red[tid] = fmaxf(red[tid], red[tid + off]);
        __syncthreads();
    }
    const float rowmax = red[0]; __syncthreads();

    float sum = 0.f;
    for (int s = tid; s <= t; s += 256) sum += __expf(Srow[s] - rowmax);
    red[tid] = sum; __syncthreads();
    for (int off = 128; off > 0; off >>= 1) {
        if (tid < off) red[tid] += red[tid + off];
        __syncthreads();
    }
    const float inv = 1.0f / red[0]; __syncthreads();

    for (int s = tid; s < T; s += 256) {
        float p = (s <= t) ? __expf(Srow[s] - rowmax) * inv : 0.0f;
        Prow[s] = (__bf16)p;
    }
}

extern "C" void kernel_launch(void* const* d_in, const int* in_sizes, int n_in,
                              void* d_out, int out_size, void* d_ws, size_t ws_size,
                              hipStream_t stream) {
    (void)in_sizes; (void)n_in; (void)out_size; (void)ws_size;
    const int B = 4, T = 2048, C = 1024;
    const int M = B * T;                      // 8192 tokens

    const float* x  = (const float*)d_in[0];
    const float* Wq = (const float*)d_in[1];
    const float* Wk = (const float*)d_in[2];
    const float* Wv = (const float*)d_in[3];
    float* out = (float*)d_out;

    // workspace (bytes): Q,K,V bf16 (16MB each) | S f32 (64MB) | P bf16 (32MB) = 144MB
    char* ws = (char*)d_ws;
    __bf16* Qb = (__bf16*)(ws);
    __bf16* Kb = (__bf16*)(ws + (size_t)16 * 1024 * 1024);
    __bf16* Vb = (__bf16*)(ws + (size_t)32 * 1024 * 1024);
    float*  Sf = (float*) (ws + (size_t)48 * 1024 * 1024);
    __bf16* Pb = (__bf16*)(ws + (size_t)112 * 1024 * 1024);

    const float qscale = 0.03125f;  // C^-0.5 = 1/32, folded into Q (exact in bf16)
    dim3 blk(256);

    // 1) QKV projections: [8192,1024] = x @ W^T (f32 in, bf16 out), 128x128 tiles
    dim3 gProj(C / 128, M / 128, 1);
    gemm_wmma<true, true, false, 0, 128, 128, 64, 32><<<gProj, blk, 0, stream>>>(
        x, Wq, Qb, C, C, C, C, 0, 0, 0, qscale);
    gemm_wmma<true, true, false, 0, 128, 128, 64, 32><<<gProj, blk, 0, stream>>>(
        x, Wk, Kb, C, C, C, C, 0, 0, 0, 1.0f);
    gemm_wmma<true, true, false, 0, 128, 128, 64, 32><<<gProj, blk, 0, stream>>>(
        x, Wv, Vb, C, C, C, C, 0, 0, 0, 1.0f);

    // 2) S = Q @ K^T per batch, causal-masked f32 [B,T,T], 128x256 tiles
    dim3 gScore(T / 256, T / 128, B);
    gemm_wmma<false, false, false, 1, 128, 256, 64, 64><<<gScore, blk, 0, stream>>>(
        Qb, Kb, Sf, C, C, C, T,
        (size_t)T * C, (size_t)T * C, (size_t)T * T, 1.0f);

    // 3) causal softmax rows -> P bf16 [B,T,T]
    softmax_causal<<<dim3(B * T), blk, 0, stream>>>(Sf, Pb, T);

    // 4) out = P @ V per batch, f32 [B,T,C], 128x256 tiles
    dim3 gOut(C / 256, T / 128, B);
    gemm_wmma<false, false, true, 2, 128, 256, 64, 64><<<gOut, blk, 0, stream>>>(
        Pb, Vb, out, T, T, C, C,
        (size_t)T * T, (size_t)T * C, (size_t)T * C, 1.0f);
}
